// CrossAttentionTransformerDiffusion_9887014715593
// MI455X (gfx1250) — compile-verified, matrix operand order checked by reading, not run on hardware
//
#include <hip/hip_runtime.h>
#include <math.h>

typedef _Float16 v16h  __attribute__((ext_vector_type(16)));
typedef _Float16 half8 __attribute__((ext_vector_type(8)));
typedef _Float16 half4 __attribute__((ext_vector_type(4)));
typedef float    v8f   __attribute__((ext_vector_type(8)));

#define BDIM 32768
#define EMB 64
#define HID 256
#define NL  6
#define TDIM 128
#define FFN 1024
#define BM 64          // rows per workgroup
#define NW 8           // waves per workgroup (256 threads, wave32)

// ---------------------------------------------------------------------------
// WMMA fragment load (ISA 7.12.2): lane l<16 holds row/col l, halves 0..7 =
// K kb..kb+7, halves 8..15 = K kb+16..kb+23; lanes 16..31 same row, K +8/+24.
// Used for A tiles (LDS) and B tiles from transposed weights WT (N x K).
// ---------------------------------------------------------------------------
__device__ __forceinline__ v16h load_frag(const _Float16* __restrict__ p0, int stride, int kb) {
  const int lane = threadIdx.x & 31;
  const int sel  = (lane >> 4) & 1;
  const int r    = lane & 15;
  const _Float16* p = p0 + r * stride + kb + sel * 8;
  half8 lo = *(const half8*)(p);
  half8 hi = *(const half8*)(p + 16);
  v16h v;
#pragma unroll
  for (int i = 0; i < 8; ++i) { v[i] = lo[i]; v[i + 8] = hi[i]; }
  return v;
}

#define WMMA_F16(a, b, c) \
  __builtin_amdgcn_wmma_f32_16x16x32_f16(false, (a), false, (b), (short)0, (c), false, false)

// ---------------------------------------------------------------------------
// GEMM over the 64-row tile. Each wave register-blocks a 64x32 output tile:
// two N-tiles (nt0 = wave + 16g, nt1 = nt0 + 8) x all four 16-row M-tiles.
// Per kb step: 4 A frags (LDS) + 2 B frags (global, one-step prefetch) feed
// 8 WMMAs from registers -> halves LDS traffic vs per-tile reloading.
// C/D layout: elem j -> row = mt*16 + j + (lane>=16 ? 8:0), col = nt*16+(lane&15)
// ---------------------------------------------------------------------------
template <int K, int NTILES, class Epi>
__device__ __forceinline__ void gemm64(const _Float16* __restrict__ A, const int As,
                                       const _Float16* __restrict__ WT, const int wave,
                                       Epi epi) {
  for (int nt0 = wave; nt0 < NTILES; nt0 += 2 * NW) {
    const int nt1 = nt0 + NW;
    const bool has2 = (nt1 < NTILES);
    const _Float16* b0p = WT + (size_t)nt0 * 16 * K;
    const _Float16* b1p = WT + (size_t)(has2 ? nt1 : nt0) * 16 * K;
    v8f acc0[4] = {v8f{}, v8f{}, v8f{}, v8f{}};
    v8f acc1[4] = {v8f{}, v8f{}, v8f{}, v8f{}};
    v16h b0 = load_frag(b0p, K, 0);
    v16h b1 = b0;
    if (has2) b1 = load_frag(b1p, K, 0);
    for (int kb = 0; kb < K; kb += 32) {
      v16h b0n = b0, b1n = b1;
      if (kb + 32 < K) {                        // prefetch next B frags
        b0n = load_frag(b0p, K, kb + 32);
        if (has2) b1n = load_frag(b1p, K, kb + 32);
      }
      v16h a0 = load_frag(A,           As, kb);
      v16h a1 = load_frag(A + 16 * As, As, kb);
      v16h a2 = load_frag(A + 32 * As, As, kb);
      v16h a3 = load_frag(A + 48 * As, As, kb);
      acc0[0] = WMMA_F16(a0, b0, acc0[0]);
      acc0[1] = WMMA_F16(a1, b0, acc0[1]);
      acc0[2] = WMMA_F16(a2, b0, acc0[2]);
      acc0[3] = WMMA_F16(a3, b0, acc0[3]);
      if (has2) {
        acc1[0] = WMMA_F16(a0, b1, acc1[0]);
        acc1[1] = WMMA_F16(a1, b1, acc1[1]);
        acc1[2] = WMMA_F16(a2, b1, acc1[2]);
        acc1[3] = WMMA_F16(a3, b1, acc1[3]);
      }
      b0 = b0n; b1 = b1n;
    }
    epi(nt0, acc0);
    if (has2) epi(nt1, acc1);
  }
}

// row LayerNorm over 256 cols of sH -> f16 sA with affine.
// lane owns 8 contiguous cols (32 lanes * 8 = 256); 8 rows per wave.
__device__ __forceinline__ void ln_rows(const float* sH, const float* __restrict__ g,
                                        const float* __restrict__ b, _Float16* sA,
                                        int wave, int lane) {
  const int c8 = lane * 8;
  for (int rr = 0; rr < 8; ++rr) {
    int r = wave * 8 + rr;
    float4 v0 = *(const float4*)(sH + r * 264 + c8);
    float4 v1 = *(const float4*)(sH + r * 264 + c8 + 4);
    float va[8] = {v0.x, v0.y, v0.z, v0.w, v1.x, v1.y, v1.z, v1.w};
    float s = 0.f, ss = 0.f;
#pragma unroll
    for (int i = 0; i < 8; ++i) { s += va[i]; ss += va[i] * va[i]; }
#pragma unroll
    for (int m = 16; m >= 1; m >>= 1) { s += __shfl_xor(s, m, 32); ss += __shfl_xor(ss, m, 32); }
    float mu   = s * (1.f / 256.f);
    float var  = ss * (1.f / 256.f) - mu * mu;
    float rstd = rsqrtf(var + 1e-5f);
    half8 o;
#pragma unroll
    for (int i = 0; i < 8; ++i)
      o[i] = (_Float16)((va[i] - mu) * rstd * g[c8 + i] + b[c8 + i]);
    *(half8*)(sA + r * 272 + c8) = o;
  }
}

// ---------------------------------------------------------------------------
// weight prep: f32 (L,K,N) -> f16 transposed (L,N,K)
// ---------------------------------------------------------------------------
__global__ void dit_prep_t(const float* __restrict__ src, _Float16* __restrict__ dst,
                           long Lb, long K, long N) {
  long total = Lb * K * N;
  for (long idx = (long)blockIdx.x * blockDim.x + threadIdx.x; idx < total;
       idx += (long)gridDim.x * blockDim.x) {
    long l = idx / (K * N); long rem = idx - l * K * N;
    long k = rem / N;       long n = rem - k * N;
    dst[l * K * N + n * K + k] = (_Float16)src[idx];
  }
}

// ---------------------------------------------------------------------------
// time-embedding kernel: sinusoid -> MLP(SiLU) -> proj(256) -> LN -> tn (f16)
// ---------------------------------------------------------------------------
__global__ __launch_bounds__(256) void dit_time(
    const float* __restrict__ tvec,
    const _Float16* __restrict__ Wt1T, const float* __restrict__ bt1,
    const _Float16* __restrict__ Wt2T, const float* __restrict__ bt2,
    const _Float16* __restrict__ WtpT, const float* __restrict__ btp,
    _Float16* __restrict__ tn) {
  extern __shared__ char smem[];
  _Float16* sT = (_Float16*)smem;                      // 64 x 144
  _Float16* sU = sT + 64 * 144;                        // 64 x 144
  float*    sV = (float*)(smem + 64 * 144 * 2 * 2);    // 64 x 264
  const int tid = threadIdx.x, wave = tid >> 5, lane = tid & 31;
  const int row0 = blockIdx.x * BM;
  const float c0 = -9.210340371976184f / 63.0f;        // -ln(10000)/(half-1)

  for (int idx = tid; idx < BM * TDIM; idx += 256) {
    int r = idx >> 7, c = idx & 127;
    float tt = tvec[row0 + r];
    int f = (c < 64) ? c : (c - 64);
    float a = tt * __expf(c0 * (float)f);
    float v = (c < 64) ? sinf(a) : cosf(a);
    sT[r * 144 + c] = (_Float16)v;
  }
  __syncthreads();
  // te1 = silu(te @ Wt1 + bt1)
  gemm64<128, 8>(sT, 144, Wt1T, wave, [&](int nt, v8f (&acc)[4]) {
    int coln = nt * 16 + (lane & 15);
    int rofs = (lane >> 4) ? 8 : 0;
    float bb = bt1[coln];
#pragma unroll
    for (int mt = 0; mt < 4; ++mt)
#pragma unroll
      for (int j = 0; j < 8; ++j) {
        float z = acc[mt][j] + bb;
        float s = z * __builtin_amdgcn_rcpf(1.f + __expf(-z));
        sU[(mt * 16 + rofs + j) * 144 + coln] = (_Float16)s;
      }
  });
  __syncthreads();
  // te2 = te1 @ Wt2 + bt2
  gemm64<128, 8>(sU, 144, Wt2T, wave, [&](int nt, v8f (&acc)[4]) {
    int coln = nt * 16 + (lane & 15);
    int rofs = (lane >> 4) ? 8 : 0;
    float bb = bt2[coln];
#pragma unroll
    for (int mt = 0; mt < 4; ++mt)
#pragma unroll
      for (int j = 0; j < 8; ++j)
        sT[(mt * 16 + rofs + j) * 144 + coln] = (_Float16)(acc[mt][j] + bb);
  });
  __syncthreads();
  // te3 = te2 @ Wtp + btp   (N=256)
  gemm64<128, 16>(sT, 144, WtpT, wave, [&](int nt, v8f (&acc)[4]) {
    int coln = nt * 16 + (lane & 15);
    int rofs = (lane >> 4) ? 8 : 0;
    float bb = btp[coln];
#pragma unroll
    for (int mt = 0; mt < 4; ++mt)
#pragma unroll
      for (int j = 0; j < 8; ++j)
        sV[(mt * 16 + rofs + j) * 264 + coln] = acc[mt][j] + bb;
  });
  __syncthreads();
  // affine-free LayerNorm -> tn (per-layer g1/be1 folded in later)
  {
    const int c8 = lane * 8;
    for (int rr = 0; rr < 8; ++rr) {
      int r = wave * 8 + rr;
      float4 v0 = *(const float4*)(sV + r * 264 + c8);
      float4 v1 = *(const float4*)(sV + r * 264 + c8 + 4);
      float va[8] = {v0.x, v0.y, v0.z, v0.w, v1.x, v1.y, v1.z, v1.w};
      float s = 0.f, ss = 0.f;
#pragma unroll
      for (int i = 0; i < 8; ++i) { s += va[i]; ss += va[i] * va[i]; }
#pragma unroll
      for (int m = 16; m >= 1; m >>= 1) { s += __shfl_xor(s, m, 32); ss += __shfl_xor(ss, m, 32); }
      float mu = s * (1.f / 256.f), var = ss * (1.f / 256.f) - mu * mu;
      float rstd = rsqrtf(var + 1e-5f);
      half8 o;
#pragma unroll
      for (int i = 0; i < 8; ++i) o[i] = (_Float16)((va[i] - mu) * rstd);
      *(half8*)(tn + (size_t)(row0 + r) * 256 + c8) = o;
    }
  }
}

// ---------------------------------------------------------------------------
// fused transformer stack: 64 rows/WG, h resident in LDS
// ---------------------------------------------------------------------------
__global__ __launch_bounds__(256) void dit_main(
    const float* __restrict__ x, const _Float16* __restrict__ tn,
    const _Float16* __restrict__ WeT, const float* __restrict__ be,
    const _Float16* __restrict__ WvcT, const float* __restrict__ bvc,
    const _Float16* __restrict__ WocT, const float* __restrict__ boc,
    const _Float16* __restrict__ WvsT, const float* __restrict__ bvs,
    const _Float16* __restrict__ WosT, const float* __restrict__ bos,
    const _Float16* __restrict__ W1T,  const float* __restrict__ b1,
    const _Float16* __restrict__ W2T,  const float* __restrict__ b2,
    const float* __restrict__ g1, const float* __restrict__ be1,
    const float* __restrict__ g2, const float* __restrict__ be2,
    const float* __restrict__ g3, const float* __restrict__ be3,
    const _Float16* __restrict__ WoutT, const float* __restrict__ bout,
    float* __restrict__ out) {
  extern __shared__ char smem[];
  _Float16* sA = (_Float16*)smem;                                  // 64 x 272 f16
  float*    sH = (float*)(smem + 64 * 272 * 2);                    // 64 x 264 f32
  _Float16* sB = (_Float16*)(smem + 64 * 272 * 2 + 64 * 264 * 4);  // 64 x 1040 f16
  const int tid = threadIdx.x, wave = tid >> 5, lane = tid & 31;
  const int row0 = blockIdx.x * BM;

  // ---- h0 = x @ We + be  (K=64) ----
  for (int idx = tid; idx < BM * 16; idx += 256) {      // lane: 4 contiguous cols
    int r = idx >> 4, c4 = (idx & 15) * 4;
    float4 xv = *(const float4*)(x + (size_t)(row0 + r) * EMB + c4);
    half4 o; o[0] = (_Float16)xv.x; o[1] = (_Float16)xv.y;
    o[2] = (_Float16)xv.z; o[3] = (_Float16)xv.w;
    *(half4*)(sA + r * 80 + c4) = o;
  }
  __syncthreads();
  gemm64<64, 16>(sA, 80, WeT, wave, [&](int nt, v8f (&acc)[4]) {
    int coln = nt * 16 + (lane & 15);
    int rofs = (lane >> 4) ? 8 : 0;
    float bb = be[coln];
#pragma unroll
    for (int mt = 0; mt < 4; ++mt)
#pragma unroll
      for (int j = 0; j < 8; ++j)
        sH[(mt * 16 + rofs + j) * 264 + coln] = acc[mt][j] + bb;
  });
  __syncthreads();

  for (int L = 0; L < NL; ++L) {
    // ---- cross-attn: sA = f16(tn * g1 + be1) ----
    const float* g1l = g1 + L * 256; const float* e1l = be1 + L * 256;
    for (int idx = tid; idx < BM * 32; idx += 256) {    // lane: 8 contiguous cols
      int r = idx >> 5, c8 = (idx & 31) * 8;
      half8 tv8 = *(const half8*)(tn + (size_t)(row0 + r) * 256 + c8);
      half8 o;
#pragma unroll
      for (int i = 0; i < 8; ++i)
        o[i] = (_Float16)((float)tv8[i] * g1l[c8 + i] + e1l[c8 + i]);
      *(half8*)(sA + r * 272 + c8) = o;
    }
    __syncthreads();
    {
      const float* bias = bvc + L * 256;
      gemm64<256, 16>(sA, 272, WvcT + (size_t)L * 65536, wave, [&](int nt, v8f (&acc)[4]) {
        int coln = nt * 16 + (lane & 15);
        int rofs = (lane >> 4) ? 8 : 0;
        float bb = bias[coln];
#pragma unroll
        for (int mt = 0; mt < 4; ++mt)
#pragma unroll
          for (int j = 0; j < 8; ++j)
            sB[(mt * 16 + rofs + j) * 272 + coln] = (_Float16)(acc[mt][j] + bb);
      });
    }
    __syncthreads();
    {
      const float* bias = boc + L * 256;
      gemm64<256, 16>(sB, 272, WocT + (size_t)L * 65536, wave, [&](int nt, v8f (&acc)[4]) {
        int coln = nt * 16 + (lane & 15);
        int rofs = (lane >> 4) ? 8 : 0;
        float bb = bias[coln];
#pragma unroll
        for (int mt = 0; mt < 4; ++mt)
#pragma unroll
          for (int j = 0; j < 8; ++j)
            sH[(mt * 16 + rofs + j) * 264 + coln] += acc[mt][j] + bb;
      });
    }
    __syncthreads();

    // ---- self-attn ----
    ln_rows(sH, g2 + L * 256, be2 + L * 256, sA, wave, lane);
    __syncthreads();
    {
      const float* bias = bvs + L * 256;
      gemm64<256, 16>(sA, 272, WvsT + (size_t)L * 65536, wave, [&](int nt, v8f (&acc)[4]) {
        int coln = nt * 16 + (lane & 15);
        int rofs = (lane >> 4) ? 8 : 0;
        float bb = bias[coln];
#pragma unroll
        for (int mt = 0; mt < 4; ++mt)
#pragma unroll
          for (int j = 0; j < 8; ++j)
            sB[(mt * 16 + rofs + j) * 272 + coln] = (_Float16)(acc[mt][j] + bb);
      });
    }
    __syncthreads();
    {
      const float* bias = bos + L * 256;
      gemm64<256, 16>(sB, 272, WosT + (size_t)L * 65536, wave, [&](int nt, v8f (&acc)[4]) {
        int coln = nt * 16 + (lane & 15);
        int rofs = (lane >> 4) ? 8 : 0;
        float bb = bias[coln];
#pragma unroll
        for (int mt = 0; mt < 4; ++mt)
#pragma unroll
          for (int j = 0; j < 8; ++j)
            sH[(mt * 16 + rofs + j) * 264 + coln] += acc[mt][j] + bb;
      });
    }
    __syncthreads();

    // ---- FFN ----
    ln_rows(sH, g3 + L * 256, be3 + L * 256, sA, wave, lane);
    __syncthreads();
    {
      const float* bias = b1 + L * FFN;
      gemm64<256, 64>(sA, 272, W1T + (size_t)L * FFN * 256, wave, [&](int nt, v8f (&acc)[4]) {
        int coln = nt * 16 + (lane & 15);
        int rofs = (lane >> 4) ? 8 : 0;
        float bb = bias[coln];
#pragma unroll
        for (int mt = 0; mt < 4; ++mt)
#pragma unroll
          for (int j = 0; j < 8; ++j) {
            float z = acc[mt][j] + bb;
            float g = 0.5f * z * (1.0f + erff(z * 0.7071067811865475f));
            sB[(mt * 16 + rofs + j) * 1040 + coln] = (_Float16)g;
          }
      });
    }
    __syncthreads();
    {
      const float* bias = b2 + L * 256;
      gemm64<1024, 16>(sB, 1040, W2T + (size_t)L * 256 * FFN, wave, [&](int nt, v8f (&acc)[4]) {
        int coln = nt * 16 + (lane & 15);
        int rofs = (lane >> 4) ? 8 : 0;
        float bb = bias[coln];
#pragma unroll
        for (int mt = 0; mt < 4; ++mt)
#pragma unroll
          for (int j = 0; j < 8; ++j)
            sH[(mt * 16 + rofs + j) * 264 + coln] += acc[mt][j] + bb;
      });
    }
    __syncthreads();
  }

  // ---- out = h @ Wout + bout (N=64) ----
  for (int idx = tid; idx < BM * 32; idx += 256) {
    int r = idx >> 5, c8 = (idx & 31) * 8;
    float4 v0 = *(const float4*)(sH + r * 264 + c8);
    float4 v1 = *(const float4*)(sH + r * 264 + c8 + 4);
    half8 o;
    o[0] = (_Float16)v0.x; o[1] = (_Float16)v0.y; o[2] = (_Float16)v0.z; o[3] = (_Float16)v0.w;
    o[4] = (_Float16)v1.x; o[5] = (_Float16)v1.y; o[6] = (_Float16)v1.z; o[7] = (_Float16)v1.w;
    *(half8*)(sA + r * 272 + c8) = o;
  }
  __syncthreads();
  gemm64<256, 4>(sA, 272, WoutT, wave, [&](int nt, v8f (&acc)[4]) {
    int coln = nt * 16 + (lane & 15);
    int rofs = (lane >> 4) ? 8 : 0;
    float bb = bout[coln];
#pragma unroll
    for (int mt = 0; mt < 4; ++mt)
#pragma unroll
      for (int j = 0; j < 8; ++j)
        out[(size_t)(row0 + mt * 16 + rofs + j) * EMB + coln] = acc[mt][j] + bb;
  });
}

// ---------------------------------------------------------------------------
extern "C" void kernel_launch(void* const* d_in, const int* in_sizes, int n_in,
                              void* d_out, int out_size, void* d_ws, size_t ws_size,
                              hipStream_t stream) {
  (void)in_sizes; (void)n_in; (void)out_size; (void)ws_size;
  // inputs in setup_inputs() dict order
  const float* x    = (const float*)d_in[0];
  const float* tv   = (const float*)d_in[1];
  const float* Wt1  = (const float*)d_in[2];  const float* bt1 = (const float*)d_in[3];
  const float* Wt2  = (const float*)d_in[4];  const float* bt2 = (const float*)d_in[5];
  const float* Wtp  = (const float*)d_in[6];  const float* btp = (const float*)d_in[7];
  const float* We   = (const float*)d_in[8];  const float* be  = (const float*)d_in[9];
  const float* Wvc  = (const float*)d_in[10]; const float* bvc = (const float*)d_in[11];
  const float* Woc  = (const float*)d_in[12]; const float* boc = (const float*)d_in[13];
  const float* Wvs  = (const float*)d_in[14]; const float* bvs = (const float*)d_in[15];
  const float* Wos  = (const float*)d_in[16]; const float* bos = (const float*)d_in[17];
  const float* W1   = (const float*)d_in[18]; const float* b1  = (const float*)d_in[19];
  const float* W2   = (const float*)d_in[20]; const float* b2  = (const float*)d_in[21];
  const float* g1   = (const float*)d_in[22]; const float* be1 = (const float*)d_in[23];
  const float* g2   = (const float*)d_in[24]; const float* be2 = (const float*)d_in[25];
  const float* g3   = (const float*)d_in[26]; const float* be3 = (const float*)d_in[27];
  const float* Wout = (const float*)d_in[28]; const float* bout= (const float*)d_in[29];
  float* out = (float*)d_out;

  // workspace carve-out (f16 transposed weights + tn), 256B aligned
  char* ws = (char*)d_ws;
  size_t off = 0;
  auto carve = [&](size_t elems) {
    size_t o = off; off = (off + elems * 2 + 255) & ~(size_t)255; return (_Float16*)(ws + o);
  };
  _Float16* WeT   = carve((size_t)256 * 64);
  _Float16* Wt1T  = carve((size_t)128 * 128);
  _Float16* Wt2T  = carve((size_t)128 * 128);
  _Float16* WtpT  = carve((size_t)256 * 128);
  _Float16* WvcT  = carve((size_t)NL * 256 * 256);
  _Float16* WocT  = carve((size_t)NL * 256 * 256);
  _Float16* WvsT  = carve((size_t)NL * 256 * 256);
  _Float16* WosT  = carve((size_t)NL * 256 * 256);
  _Float16* W1T   = carve((size_t)NL * 256 * FFN);
  _Float16* W2T   = carve((size_t)NL * FFN * 256);
  _Float16* WoutT = carve((size_t)64 * 256);
  _Float16* tn    = carve((size_t)BDIM * 256);

  auto prep = [&](const float* src, _Float16* dst, long Lb, long K, long N) {
    long total = Lb * K * N;
    int grid = (int)((total + 255) / 256);
    if (grid > 8192) grid = 8192;
    dit_prep_t<<<grid, 256, 0, stream>>>(src, dst, Lb, K, N);
  };
  prep(We, WeT, 1, 64, 256);
  prep(Wt1, Wt1T, 1, 128, 128);
  prep(Wt2, Wt2T, 1, 128, 128);
  prep(Wtp, WtpT, 1, 128, 256);
  prep(Wvc, WvcT, NL, 256, 256);
  prep(Woc, WocT, NL, 256, 256);
  prep(Wvs, WvsT, NL, 256, 256);
  prep(Wos, WosT, NL, 256, 256);
  prep(W1, W1T, NL, 256, FFN);
  prep(W2, W2T, NL, FFN, 256);
  prep(Wout, WoutT, 1, 256, 64);

  const int smem_time = 64 * 144 * 2 * 2 + 64 * 264 * 4;                // ~102 KB
  const int smem_main = 64 * 272 * 2 + 64 * 264 * 4 + 64 * 1040 * 2;    // ~230 KB
  (void)hipFuncSetAttribute((const void*)dit_time,
      hipFuncAttributeMaxDynamicSharedMemorySize, smem_time);
  (void)hipFuncSetAttribute((const void*)dit_main,
      hipFuncAttributeMaxDynamicSharedMemorySize, smem_main);

  dit_time<<<BDIM / BM, 256, smem_time, stream>>>(tv, Wt1T, bt1, Wt2T, bt2, WtpT, btp, tn);

  dit_main<<<BDIM / BM, 256, smem_main, stream>>>(
      x, tn, WeT, be,
      WvcT, bvc, WocT, boc, WvsT, bvs, WosT, bos,
      W1T, b1, W2T, b2,
      g1, be1, g2, be2, g3, be3,
      WoutT, bout, out);
}
